// GATv2layer_19378892439975
// MI455X (gfx1250) — matrix operand order repaired and checked by reading, not compiled, not run on hardware
//
#include <hip/hip_runtime.h>
#include <hip/hip_bf16.h>

#define SLOPE 0.01f

typedef _Float16 h2    __attribute__((ext_vector_type(2)));
typedef __bf16   v16bf __attribute__((ext_vector_type(16)));
typedef float    v8f   __attribute__((ext_vector_type(8)));
typedef unsigned int u32x4 __attribute__((ext_vector_type(4)));
typedef int          i32x8 __attribute__((ext_vector_type(8)));
typedef int          i32x4 __attribute__((ext_vector_type(4)));

__device__ __forceinline__ unsigned short f2bf(float f) {
    union { float f; unsigned int u; } v; v.f = f;
    unsigned int u = v.u;
    unsigned int r = (u + 0x7fffu + ((u >> 16) & 1u)) >> 16;
    return (unsigned short)r;
}

// LDS byte offset of a __shared__ object: flat LDS aperture puts the
// workgroup-relative offset in addr[31:0] (ISA 10.2 aperture rules).
__device__ __forceinline__ unsigned int lds_off(const void* p) {
    return (unsigned int)(unsigned long long)p;
}

// Issue a 2D TENSOR_LOAD_TO_LDS: tile [rows x cols] of elemSize-byte data,
// global row pitch = pitchElems elements, into contiguous LDS at ldsaddr.
// D# bitfields per CDNA5 ISA 8.3/8.4 (group0/group1); groups 2/3 unused (2D).
// 6-arg builtin form (clang-23 / therock-10.0 headers).
__device__ __forceinline__ void tdm_load_2d(unsigned int ldsaddr,
                                            unsigned long long gaddr,
                                            unsigned int dszCode,   // 0=1B,1=2B,2=4B
                                            unsigned int tensorDim0,
                                            unsigned int pitchElems,
                                            unsigned int tileCols,
                                            unsigned int tileRows) {
    u32x4 g0 = {
        1u,                                              // count=1, no gather
        ldsaddr,                                         // lds_addr [63:32]
        (unsigned int)(gaddr & 0xffffffffu),             // global_addr lo
        (unsigned int)((gaddr >> 32) & 0x1ffffffu) | (2u << 30)  // hi + type=2
    };
    i32x8 g1 = {
        (int)(dszCode << 16),            // [17:16] data_size
        (int)((tensorDim0 & 0xffffu) << 16),   // tensor_dim0[15:0] @ [63:48]
        (int)(((tensorDim0 >> 16) & 0xffffu) | ((tileRows & 0xffffu) << 16)),
                                         // tensor_dim0 hi + tensor_dim1[15:0]
        (int)((tileCols & 0xffffu) << 16),     // tensor_dim1 hi=0, tile_dim0
        (int)(tileRows & 0xffffu),             // tile_dim1 ; tile_dim2=0
        (int)pitchElems,                       // tensor_dim0_stride lo
        0, 0                                   // stride hi, dim1_stride (2D)
    };
    i32x4 gz4 = {0, 0, 0, 0};
    i32x8 gz8 = {0, 0, 0, 0, 0, 0, 0, 0};
    __builtin_amdgcn_tensor_load_to_lds(g0, g1, gz4, gz4, gz8, 0);
}

// ---------------------------------------------------------------------------
// K1: x1 = x@w1 (f16), x2 = x@w2 (f16), xbT[b][d][n] = bf16(x[b][n][d])
// ---------------------------------------------------------------------------
__global__ void __launch_bounds__(64) gat_proj_kernel(
    const float* __restrict__ x, const float* __restrict__ w1,
    const float* __restrict__ w2,
    _Float16* __restrict__ x1h, _Float16* __restrict__ x2h,
    unsigned short* __restrict__ xbT)
{
    const int row = blockIdx.x;        // b*1024 + n
    const int b = row >> 10, n = row & 1023;
    const int t = threadIdx.x;
    __shared__ float xr[128];
    xr[t]      = x[(size_t)row * 128 + t];
    xr[t + 64] = x[(size_t)row * 128 + t + 64];
    __syncthreads();
    float s1 = 0.f, s2 = 0.f;
    #pragma unroll 8
    for (int k = 0; k < 128; ++k) {
        float xv = xr[k];
        s1 = fmaf(xv, w1[k * 64 + t], s1);
        s2 = fmaf(xv, w2[k * 64 + t], s2);
    }
    x1h[(size_t)row * 64 + t] = (_Float16)s1;
    x2h[(size_t)row * 64 + t] = (_Float16)s2;
    xbT[((size_t)b * 128 + t)      * 1024 + n] = f2bf(xr[t]);
    xbT[((size_t)b * 128 + t + 64) * 1024 + n] = f2bf(xr[t + 64]);
}

// ---------------------------------------------------------------------------
// K2: fused scores + masked softmax + aggregation (A@x) via bf16 WMMA,
// with TDM (tensor_load_to_lds) staging of the bf16 feature tile and the
// adjacency-mask tile.  grid = B*(N/16) = 256 blocks, 256 threads (8 waves).
// ---------------------------------------------------------------------------
__global__ void __launch_bounds__(256) gat_attn_kernel(
    const float* __restrict__ Amask,        // [B,N,N]
    const float* __restrict__ a_vec,        // [64]
    const _Float16* __restrict__ x1h,       // [B*N,64]
    const _Float16* __restrict__ x2h,       // [B*N,64]
    const unsigned short* __restrict__ xbT, // [B,128,N] bf16 bits
    float* __restrict__ oag)                // [B*N,128]
{
    const int b     = blockIdx.x >> 6;
    const int ibase = (blockIdx.x & 63) * 16;
    const int tid = threadIdx.x;
    const int w   = tid >> 5;      // wave 0..7
    const int l   = tid & 31;      // lane
    const int jl  = l & 15;
    const int hi  = l >> 4;        // half-wave select

    __shared__ __attribute__((aligned(32))) _Float16       x1s[16][64];   // 2 KB
    __shared__ __attribute__((aligned(32))) _Float16       x2s[128][64];  // 16 KB
    __shared__ __attribute__((aligned(32))) unsigned short xbs[128][128]; // 32 KB bf16
    __shared__ __attribute__((aligned(32))) unsigned short Ebuf[16][128]; // 4 KB bf16
    __shared__ __attribute__((aligned(32))) float          mbuf[16][128]; // 8 KB mask
    __shared__ _Float16 as2[64];
    __shared__ float    rowsum[16];

    if (tid < 16) rowsum[tid] = 0.f;
    if (tid < 64) as2[tid] = (_Float16)a_vec[tid];
    for (int k = tid; k < 16 * 64; k += 256) {
        int i = k >> 6, d = k & 63;
        x1s[i][d] = x1h[(size_t)(b * 1024 + ibase + i) * 64 + d];
    }
    __syncthreads();

    v8f acc = {0.f, 0.f, 0.f, 0.f, 0.f, 0.f, 0.f, 0.f};   // (i=r+8*hi, d=w*16+jl)
    float rs[8];
    #pragma unroll
    for (int r = 0; r < 8; ++r) rs[r] = 0.f;

    const h2 cneg = {(_Float16)(SLOPE - 1.f), (_Float16)(SLOPE - 1.f)};
    const h2 zero = {(_Float16)0.f, (_Float16)0.f};

    for (int t = 0; t < 8; ++t) {
        const int j0 = t * 128;
        __syncthreads();   // previous chunk fully consumed

        // ---- TDM: async 2D tile DMAs into LDS -------------------------------
        if (w == 0) {
            // bf16 feature tile: 128 d-rows x 128 j-cols, global pitch 1024
            unsigned long long ga = (unsigned long long)(xbT + (size_t)b * 128 * 1024 + j0);
            tdm_load_2d(lds_off(&xbs[0][0]), ga, /*2B*/1u,
                        1024u, 1024u, 128u, 128u);
        } else if (w == 1) {
            // mask tile: 16 i-rows x 128 j-cols f32, global pitch 1024
            unsigned long long ga =
                (unsigned long long)(Amask + ((size_t)b * 1024 + ibase) * 1024 + j0);
            tdm_load_2d(lds_off(&mbuf[0][0]), ga, /*4B*/2u,
                        1024u, 1024u, 128u, 16u);
        }
        // ---- cooperative load of the f16 x2 chunk (contiguous 16 KB) -------
        {
            const unsigned int* gx2 =
                (const unsigned int*)(x2h + (size_t)(b * 1024 + j0) * 64);
            unsigned int* sx2 = (unsigned int*)&x2s[0][0];
            for (int k = tid; k < 128 * 32; k += 256) sx2[k] = gx2[k];
        }
        // prefetch next chunk's mask rows into L2 while we compute
        if (t < 7 && tid < 16)
            __builtin_prefetch(Amask + ((size_t)b * 1024 + ibase + tid) * 1024 + j0 + 128, 0, 0);
        if (w < 2) __builtin_amdgcn_s_wait_tensorcnt(0);
        __syncthreads();

        // ---- pairwise GATv2 scores for this wave's 16x16 tile, packed f16 --
        const int jc = w * 16 + jl;                 // chunk-local column
        h2 sacc[8];
        #pragma unroll
        for (int r = 0; r < 8; ++r) sacc[r] = zero;
        {
            const h2* x2row = (const h2*)&x2s[jc][0];
            const h2* arow  = (const h2*)&as2[0];
            for (int dk = 0; dk < 32; ++dk) {
                h2 av = arow[dk];
                h2 xv = x2row[dk];
                #pragma unroll
                for (int r = 0; r < 8; ++r) {
                    h2 ts = ((const h2*)&x1s[hi * 8 + r][0])[dk] + xv;   // x1_i + x2_j
                    h2 mn = __builtin_elementwise_min(ts, zero);
                    h2 lr = ts + mn * cneg;                              // leakyrelu
                    sacc[r] += av * lr;
                }
            }
        }
        // tanh clamp, exp, LDS mask, rowsum, bf16 store to Ebuf
        #pragma unroll
        for (int r = 0; r < 8; ++r) {
            float s = (float)sacc[r][0] + (float)sacc[r][1];
            s = tanhf(s * 0.125f) * 8.0f;
            float e = __expf(s) * mbuf[hi * 8 + r][jc];
            rs[r] += e;
            Ebuf[hi * 8 + r][jc] = f2bf(e);
        }
        __syncthreads();

        // ---- WMMA: out[16i x 16d_w] += E[16 x 128] @ xbs[128 x 128] (K=j) ---
        // A frag (16-bit 16x32): lane row = jl, K base = hi*8; VGPR0-3 = K base..base+7,
// VGPR4-7 = K base+16..+23.  B frag (32x16): lane col d = w*16+jl,
// same K order as A (K base = hi*8, then base+16) in the d-major LDS row.
        {
            const int drow = w * 16 + jl;
            #pragma unroll
            for (int kk = 0; kk < 4; ++kk) {
                union { uint4 q[2]; v16bf v; } ua, ub;
                ua.q[0] = *(const uint4*)&Ebuf[jl][kk * 32 + hi * 8];
                ua.q[1] = *(const uint4*)&Ebuf[jl][kk * 32 + hi * 8 + 16];
                ub.q[0] = *(const uint4*)&xbs[drow][kk * 32 + hi * 8];
                ub.q[1] = *(const uint4*)&xbs[drow][kk * 32 + hi * 8 + 16];
                acc = __builtin_amdgcn_wmma_f32_16x16x32_bf16(
                    false, ua.v, false, ub.v, (short)0, acc, false, false);
            }
        }
    }

    // ---- cross-lane rowsum reduction, then normalize + write 16x128 tile ----
    #pragma unroll
    for (int r = 0; r < 8; ++r) atomicAdd(&rowsum[hi * 8 + r], rs[r]);
    __syncthreads();
    #pragma unroll
    for (int r = 0; r < 8; ++r) {
        const int i = hi * 8 + r;
        float v = acc[r] / rowsum[i];
        oag[((size_t)(b * 1024 + ibase + i)) * 128 + w * 16 + jl] = v;
    }
}

// ---------------------------------------------------------------------------
// K3: out = leakyrelu(oag @ lin_w.T) ; grid = B*N blocks, 64 threads
// ---------------------------------------------------------------------------
__global__ void __launch_bounds__(64) gat_out_kernel(
    const float* __restrict__ oag, const float* __restrict__ lin_w,
    float* __restrict__ out)
{
    const int row = blockIdx.x;
    const int t = threadIdx.x;
    __shared__ float orow[128];
    orow[t]      = oag[(size_t)row * 128 + t];
    orow[t + 64] = oag[(size_t)row * 128 + t + 64];
    __syncthreads();
    float s = 0.f;
    #pragma unroll 8
    for (int k = 0; k < 128; ++k) s = fmaf(orow[k], lin_w[t * 128 + k], s);
    s = (s >= 0.f) ? s : SLOPE * s;
    out[(size_t)row * 64 + t] = s;
}

// ---------------------------------------------------------------------------
extern "C" void kernel_launch(void* const* d_in, const int* in_sizes, int n_in,
                              void* d_out, int out_size, void* d_ws, size_t ws_size,
                              hipStream_t stream) {
    const float* x     = (const float*)d_in[0];   // [4,1024,128]
    const float* Amask = (const float*)d_in[1];   // [4,1024,1024]
    const float* w1    = (const float*)d_in[2];   // [128,64]
    const float* w2    = (const float*)d_in[3];   // [128,64]
    const float* a     = (const float*)d_in[4];   // [64]
    const float* lin_w = (const float*)d_in[5];   // [64,128]
    char* ws = (char*)d_ws;
    _Float16*       x1h = (_Float16*)(ws);                  // 512 KB
    _Float16*       x2h = (_Float16*)(ws + 524288);         // 512 KB
    unsigned short* xbT = (unsigned short*)(ws + 1048576);  // 1 MB
    float*          oag = (float*)(ws + 2097152);           // 2 MB
    (void)in_sizes; (void)n_in; (void)out_size; (void)ws_size;

    gat_proj_kernel<<<4096, 64, 0, stream>>>(x, w1, w2, x1h, x2h, xbT);
    gat_attn_kernel<<<256, 256, 0, stream>>>(Amask, a, x1h, x2h, xbT, oag);
    gat_out_kernel<<<4096, 64, 0, stream>>>(oag, lin_w, (float*)d_out);
}
